// BinomialLoss_57277683859994
// MI455X (gfx1250) — matrix operand order repaired and checked by reading, not compiled, not run on hardware
//
#include <hip/hip_runtime.h>
#include <hip/hip_bf16.h>

// ---------------------------------------------------------------------------
// BinomialLoss on MI455X (gfx1250):
//   sim = X @ X^T via v_wmma_f32_16x16x32_bf16.
//   Double-buffered LDS tiles filled with GLOBAL_LOAD_ASYNC_TO_LDS_B128
//   (ASYNCcnt-tracked, no VGPR staging). Steady-state K-loop is branch-free
//   (last iteration peeled). Fused masked-softplus epilogue, per-row atomics,
//   tiny final reduction -> scalar loss.
// ---------------------------------------------------------------------------

typedef __attribute__((ext_vector_type(16))) __bf16         v16bf;
typedef __attribute__((ext_vector_type(8)))  float          v8f;
typedef __attribute__((ext_vector_type(8)))  unsigned short ushort8x;

#define TILE_M   128
#define TILE_N   128
#define TILE_K   32
#define LDT      40      // LDS row stride in ushorts (80B, 16B-aligned, padded)

union Frag16 {
    v16bf    v;
    ushort8x h[2];
};

__device__ __forceinline__ unsigned short f32_to_bf16_rne(float f) {
    unsigned int u = __float_as_uint(f);
    unsigned int r = (u + 0x7FFFu + ((u >> 16) & 1u)) >> 16;
    return (unsigned short)r;
}

__device__ __forceinline__ float softplusf(float x) {
    // stable log(1+exp(x)) = max(x,0) + log(1+exp(-|x|))
    return fmaxf(x, 0.0f) + __logf(1.0f + __expf(-fabsf(x)));
}

// Async global -> LDS copy, 16 bytes per lane. LDS destination passed as a
// 32-bit LDS byte offset (low 32 bits of a generic LDS address).
__device__ __forceinline__ void async_copy_b128(unsigned lds_off,
                                                const unsigned short* g) {
    asm volatile("global_load_async_to_lds_b128 %0, %1, off"
                 :: "v"(lds_off), "v"(g)
                 : "memory");
}
__device__ __forceinline__ void wait_asynccnt_4() {
    asm volatile("s_wait_asynccnt 0x4" ::: "memory");
}
__device__ __forceinline__ void wait_asynccnt_0() {
    asm volatile("s_wait_asynccnt 0x0" ::: "memory");
}

// --- kernel 1: f32 -> bf16 bits (RNE) ---------------------------------------
__global__ void cvt_bf16_kernel(const float* __restrict__ x,
                                unsigned short* __restrict__ y, int cnt) {
    for (int i = blockIdx.x * blockDim.x + threadIdx.x; i < cnt;
         i += gridDim.x * blockDim.x)
        y[i] = f32_to_bf16_rne(x[i]);
}

// --- kernel 2: zero the per-row accumulators --------------------------------
__global__ void zero_kernel(float* __restrict__ p, int cnt) {
    for (int i = blockIdx.x * blockDim.x + threadIdx.x; i < cnt;
         i += gridDim.x * blockDim.x)
        p[i] = 0.0f;
}

// --- kernel 3: fused WMMA GEMM + binomial-loss epilogue ---------------------
__global__ __launch_bounds__(256)
void binomial_gemm_kernel(const unsigned short* __restrict__ Xb,  // [n,d] bf16 bits
                          const int* __restrict__ tgt,            // [n]
                          float* __restrict__ pos_sum, float* __restrict__ neg_sum,
                          float* __restrict__ pos_cnt, float* __restrict__ neg_cnt,
                          int n, int d) {
    __shared__ __align__(16) unsigned short sA[2][TILE_M * LDT];  // 2 x 10 KB
    __shared__ __align__(16) unsigned short sB[2][TILE_N * LDT];  // 2 x 10 KB

    const int tid  = threadIdx.x;
    const int wave = tid >> 5;            // 0..7
    const int lane = tid & 31;
    const int lh   = lane & 15;           // lane within half
    const int half = lane >> 4;           // 0 or 1

    const int wm = wave >> 1;             // 0..3  -> 32-row strip
    const int wn = wave & 1;              // 0..1  -> 64-col strip

    const int rowBlock = blockIdx.x * TILE_M;
    const int colBlock = blockIdx.y * TILE_N;
    const int rowBase  = rowBlock + wm * 32;
    const int colBase  = colBlock + wn * 64;

    // cooperative-load coordinates: 256 thr cover 128 rows x 32 k (16 shorts each)
    const int ldRow   = tid >> 1;          // 0..127
    const int ldChunk = (tid & 1) * 16;    // 0 or 16

    v8f acc[2][4];
#pragma unroll
    for (int tm = 0; tm < 2; ++tm)
#pragma unroll
        for (int tn = 0; tn < 4; ++tn)
            acc[tm][tn] = (v8f){0.f, 0.f, 0.f, 0.f, 0.f, 0.f, 0.f, 0.f};

    // A-operand ISA K-layout: half 0 holds k{0..7,16..23}, half 1 holds k{8..15,24..31}
    const int aOff = half ? 8 : 0;
    // B-operand: same K-layout as A: half 0 holds k{0..7,16..23}, half 1 holds k{8..15,24..31}
    const int bOff = half ? 8 : 0;

    const unsigned short* gA = Xb + (size_t)(rowBlock + ldRow) * d + ldChunk;
    const unsigned short* gB = Xb + (size_t)(colBlock + ldRow) * d + ldChunk;

    // Issue one stage of async global->LDS copies (4 x b128 per thread).
    auto issue_stage = [&](int buf, int kb) {
        const unsigned short* ga = gA + kb;
        const unsigned short* gb = gB + kb;
        const unsigned la = (unsigned)(size_t)&sA[buf][ldRow * LDT + ldChunk];
        const unsigned lb = (unsigned)(size_t)&sB[buf][ldRow * LDT + ldChunk];
        async_copy_b128(la,      ga);
        async_copy_b128(la + 16, ga + 8);
        async_copy_b128(lb,      gb);
        async_copy_b128(lb + 16, gb + 8);
        if (kb + TILE_K < d) {                 // global_prefetch_b8 next tile
            __builtin_prefetch(ga + TILE_K, 0, 1);
            __builtin_prefetch(gb + TILE_K, 0, 1);
        }
    };

    // Load fragments from LDS buffer `buf` and run the 8 WMMAs of one K-step.
    auto compute_stage = [&](int buf) {
        Frag16 af[2];
#pragma unroll
        for (int tm = 0; tm < 2; ++tm) {
            const int r = (wm * 32 + tm * 16 + lh) * LDT;
            af[tm].h[0] = *(const ushort8x*)&sA[buf][r + aOff];
            af[tm].h[1] = *(const ushort8x*)&sA[buf][r + aOff + 16];
        }
        Frag16 bfr[4];
#pragma unroll
        for (int tn = 0; tn < 4; ++tn) {
            const int c = (wn * 64 + tn * 16 + lh) * LDT;
            bfr[tn].h[0] = *(const ushort8x*)&sB[buf][c + bOff];
            bfr[tn].h[1] = *(const ushort8x*)&sB[buf][c + bOff + 16];
        }
#pragma unroll
        for (int tm = 0; tm < 2; ++tm)
#pragma unroll
            for (int tn = 0; tn < 4; ++tn)
                acc[tm][tn] = __builtin_amdgcn_wmma_f32_16x16x32_bf16(
                    false, af[tm].v, false, bfr[tn].v,
                    (short)0, acc[tm][tn], false, false);
    };

    const int nk = d / TILE_K;
    issue_stage(0, 0);

    // Steady state: branch-free software pipeline.
    for (int i = 0; i < nk - 1; ++i) {
        issue_stage((i + 1) & 1, (i + 1) * TILE_K);
        wait_asynccnt_4();       // async loads land in order: stage i is done
        __syncthreads();         // all waves' tile data visible in LDS
        compute_stage(i & 1);
        __syncthreads();         // done reading buf; next iter refills it
    }
    // Peeled final iteration.
    wait_asynccnt_0();
    __syncthreads();
    compute_stage((nk - 1) & 1);

    // ---- fused epilogue: masked softplus, per-row reduce, atomics ----
    int tcol[4];
#pragma unroll
    for (int tn = 0; tn < 4; ++tn)
        tcol[tn] = tgt[colBase + tn * 16 + lh];

#pragma unroll
    for (int tm = 0; tm < 2; ++tm) {
#pragma unroll
        for (int r = 0; r < 8; ++r) {
            // C layout: VGPR r -> row r (lanes 0-15) / row 8+r (lanes 16-31)
            const int grow = rowBase + tm * 16 + half * 8 + r;
            const int tr   = tgt[grow];
            float psum = 0.f, nsum = 0.f, pcnt = 0.f, ncnt = 0.f;
#pragma unroll
            for (int tn = 0; tn < 4; ++tn) {
                const float s    = acc[tm][tn][r];
                const bool  same = (tr == tcol[tn]);
                const bool  posm = same && (s < 1.0f);
                const float sp_p = softplusf(-2.0f * (s - 0.5f));
                const float sp_n = softplusf(25.0f * (s - 0.5f));
                psum += posm ? sp_p : 0.0f;
                pcnt += posm ? 1.0f : 0.0f;
                nsum += same ? 0.0f : sp_n;
                ncnt += same ? 0.0f : 1.0f;
            }
            // reduce over the 16 lanes of this half (low 4 lane-id bits)
#pragma unroll
            for (int m = 1; m < 16; m <<= 1) {
                psum += __shfl_xor(psum, m);
                nsum += __shfl_xor(nsum, m);
                pcnt += __shfl_xor(pcnt, m);
                ncnt += __shfl_xor(ncnt, m);
            }
            if (lh == 0) {
                atomicAdd(&pos_sum[grow], psum);
                atomicAdd(&neg_sum[grow], nsum);
                atomicAdd(&pos_cnt[grow], pcnt);
                atomicAdd(&neg_cnt[grow], ncnt);
            }
        }
    }
}

// --- kernel 4: per-row means -> scalar loss ---------------------------------
__global__ void finalize_kernel(const float* __restrict__ ps,
                                const float* __restrict__ ns,
                                const float* __restrict__ pc,
                                const float* __restrict__ nc,
                                float* __restrict__ out, int n) {
    float local = 0.f;
    for (int r = threadIdx.x; r < n; r += blockDim.x) {
        const float pcv = pc[r], ncv = nc[r];
        const float pm = pcv > 0.f ? ps[r] / pcv : 0.f;
        const float nm = ncv > 0.f ? ns[r] / ncv : 0.f;
        local += pm + nm;
    }
#pragma unroll
    for (int off = 16; off > 0; off >>= 1) local += __shfl_down(local, off);
    __shared__ float wsum[8];
    const int lane = threadIdx.x & 31, w = threadIdx.x >> 5;
    if (lane == 0) wsum[w] = local;
    __syncthreads();
    if (threadIdx.x == 0) {
        float t = 0.f;
#pragma unroll
        for (int i = 0; i < 8; ++i) t += wsum[i];
        out[0] = t / (float)n;
    }
}

// ---------------------------------------------------------------------------
extern "C" void kernel_launch(void* const* d_in, const int* in_sizes, int n_in,
                              void* d_out, int out_size, void* d_ws, size_t ws_size,
                              hipStream_t stream) {
    const float* X   = (const float*)d_in[0];
    const int*   tgt = (const int*)d_in[1];
    const int n = in_sizes[1];              // 4096
    const int d = in_sizes[0] / n;          // 512

    unsigned short* Xb = (unsigned short*)d_ws;
    float* pos_sum = (float*)((char*)d_ws + (size_t)n * d * sizeof(unsigned short));
    float* neg_sum = pos_sum + n;
    float* pos_cnt = neg_sum + n;
    float* neg_cnt = pos_cnt + n;

    const int cnt = n * d;
    int grid = (cnt + 1023) / 1024;
    if (grid > 4096) grid = 4096;
    cvt_bf16_kernel<<<grid, 1024, 0, stream>>>(X, Xb, cnt);
    zero_kernel<<<(4 * n + 255) / 256, 256, 0, stream>>>(pos_sum, 4 * n);

    dim3 g(n / TILE_M, n / TILE_N);
    binomial_gemm_kernel<<<g, 256, 0, stream>>>(Xb, tgt, pos_sum, neg_sum,
                                                pos_cnt, neg_cnt, n, d);

    finalize_kernel<<<1, 256, 0, stream>>>(pos_sum, neg_sum, pos_cnt, neg_cnt,
                                           (float*)d_out, n);
}
